// MultiHeadAttention_1984274890961
// MI455X (gfx1250) — compile-verified, matrix operand order checked by reading, not run on hardware
//
#include <hip/hip_runtime.h>
#include <math.h>

// ---------------------------------------------------------------------------
// Types for CDNA5 WMMA (wave32, 16x16x32 bf16 -> f32)
// ---------------------------------------------------------------------------
typedef __attribute__((ext_vector_type(16))) __bf16 v16bf;
typedef __attribute__((ext_vector_type(8)))  __bf16 v8bf;
typedef __attribute__((ext_vector_type(8)))  float  v8f;

#define B_   2
#define N_   2048
#define C_   1024
#define H_   16
#define D_   64
#define M_   (B_ * N_)        // 4096 rows of x
#define QKVN (3 * H_ * D_)    // 3072
#define SCALE_ 0.125f         // 64^-0.5

__device__ __forceinline__ v8f zero8() {
    v8f z = {0.f, 0.f, 0.f, 0.f, 0.f, 0.f, 0.f, 0.f};
    return z;
}

__device__ __forceinline__ v8f wmma_bf16(v16bf a, v16bf b, v8f c) {
    return __builtin_amdgcn_wmma_f32_16x16x32_bf16(
        /*neg_a=*/false, a, /*neg_b=*/false, b,
        /*c_mod=*/(short)0, c, /*reuse_a=*/false, /*reuse_b=*/false);
}

// Load a 16x32 A/B tile from a row-major (contiguous-K) bf16 source.
// Lane L<16 : row L, K {0..7, 16..23}; lane L>=16 : row L-16, K {8..15, 24..31}
__device__ __forceinline__ v16bf load_tile16x32(const __bf16* __restrict__ base,
                                                int ld, int lane) {
    const int half = lane >> 4;
    const int r    = lane & 15;
    const __bf16* p = base + (size_t)r * ld + half * 8;
    v8bf lo = *(const v8bf*)(p);
    v8bf hi = *(const v8bf*)(p + 16);
    return __builtin_shufflevector(lo, hi, 0, 1, 2, 3, 4, 5, 6, 7,
                                           8, 9, 10, 11, 12, 13, 14, 15);
}

// ---------------------------------------------------------------------------
// Software-pipelined 32x64 wave-tile GEMM core over K = C_ (bf16 x bf16 -> f32)
// aB: A origin (row m0, row-stride C_); bB: B^T origin (row n0, row-stride C_)
// acc[mi][s] : mi = M subtile (0..1), s = N subtile (0..3)
// ---------------------------------------------------------------------------
__device__ __forceinline__ void gemm_core_32x64(const __bf16* __restrict__ aB,
                                                const __bf16* __restrict__ bB,
                                                int lane, v8f acc[2][4]) {
    v16bf a0 = load_tile16x32(aB, C_, lane);
    v16bf a1 = load_tile16x32(aB + 16 * C_, C_, lane);
    v16bf b0 = load_tile16x32(bB + 0 * 16 * C_, C_, lane);
    v16bf b1 = load_tile16x32(bB + 1 * 16 * C_, C_, lane);
    v16bf b2 = load_tile16x32(bB + 2 * 16 * C_, C_, lane);
    v16bf b3 = load_tile16x32(bB + 3 * 16 * C_, C_, lane);

    for (int kk = 0; kk < C_ - 32; kk += 32) {
        const int kn = kk + 32;
        // Issue next K-step loads before consuming current tiles (double buffer)
        v16bf na0 = load_tile16x32(aB + kn, C_, lane);
        v16bf na1 = load_tile16x32(aB + 16 * C_ + kn, C_, lane);
        v16bf nb0 = load_tile16x32(bB + 0 * 16 * C_ + kn, C_, lane);
        v16bf nb1 = load_tile16x32(bB + 1 * 16 * C_ + kn, C_, lane);
        v16bf nb2 = load_tile16x32(bB + 2 * 16 * C_ + kn, C_, lane);
        v16bf nb3 = load_tile16x32(bB + 3 * 16 * C_ + kn, C_, lane);

        acc[0][0] = wmma_bf16(a0, b0, acc[0][0]);
        acc[1][0] = wmma_bf16(a1, b0, acc[1][0]);
        acc[0][1] = wmma_bf16(a0, b1, acc[0][1]);
        acc[1][1] = wmma_bf16(a1, b1, acc[1][1]);
        acc[0][2] = wmma_bf16(a0, b2, acc[0][2]);
        acc[1][2] = wmma_bf16(a1, b2, acc[1][2]);
        acc[0][3] = wmma_bf16(a0, b3, acc[0][3]);
        acc[1][3] = wmma_bf16(a1, b3, acc[1][3]);

        a0 = na0; a1 = na1;
        b0 = nb0; b1 = nb1; b2 = nb2; b3 = nb3;
    }
    // Final K-step
    acc[0][0] = wmma_bf16(a0, b0, acc[0][0]);
    acc[1][0] = wmma_bf16(a1, b0, acc[1][0]);
    acc[0][1] = wmma_bf16(a0, b1, acc[0][1]);
    acc[1][1] = wmma_bf16(a1, b1, acc[1][1]);
    acc[0][2] = wmma_bf16(a0, b2, acc[0][2]);
    acc[1][2] = wmma_bf16(a1, b2, acc[1][2]);
    acc[0][3] = wmma_bf16(a0, b3, acc[0][3]);
    acc[1][3] = wmma_bf16(a1, b3, acc[1][3]);
}

// ---------------------------------------------------------------------------
// Kernel 0: fp32 -> bf16 elementwise convert
// ---------------------------------------------------------------------------
__global__ __launch_bounds__(256) void mha_cvt_bf16(const float* __restrict__ in,
                                                    __bf16* __restrict__ out, int n) {
    int i = blockIdx.x * blockDim.x + threadIdx.x;
    if (i < n) out[i] = (__bf16)in[i];
}

// ---------------------------------------------------------------------------
// Kernel 1: QKV projection. qkv = x @ w_qkv^T + b_qkv, scattered to
// q/k/v [B,H,N,D] bf16, q pre-scaled by SCALE.
// Block = 128 threads (4 waves). Block tile 128x64, wave tile 32x64.
// ---------------------------------------------------------------------------
__global__ __launch_bounds__(128) void mha_qkv_gemm(
    const __bf16* __restrict__ xb,     // [M_, C_]
    const __bf16* __restrict__ wb,     // [QKVN, C_] (i.e. B^T row-major)
    const float*  __restrict__ bqkv,   // [QKVN]
    __bf16* __restrict__ q, __bf16* __restrict__ k, __bf16* __restrict__ v) {
    const int wave = threadIdx.x >> 5;
    const int lane = threadIdx.x & 31;
    const int m0   = blockIdx.x * 128 + wave * 32;
    const int n0   = blockIdx.y * 64;

    v8f acc[2][4] = {{zero8(), zero8(), zero8(), zero8()},
                     {zero8(), zero8(), zero8(), zero8()}};
    gemm_core_32x64(xb + (size_t)m0 * C_, wb + (size_t)n0 * C_, lane, acc);

    // Epilogue: bias, scale (q only), scatter to [B,H,N,D]
    const int col = lane & 15;
    const int rb  = (lane < 16) ? 0 : 8;
#pragma unroll
    for (int s = 0; s < 4; ++s) {
        const int n_g   = n0 + s * 16 + col;
        const int which = n_g >> 10;          // 0=q, 1=k, 2=v
        const int h     = (n_g >> 6) & 15;
        const int d     = n_g & 63;
        const float bias = bqkv[n_g];
        const float scl  = (which == 0) ? SCALE_ : 1.0f;
        __bf16* dst = (which == 0) ? q : (which == 1) ? k : v;
#pragma unroll
        for (int mi = 0; mi < 2; ++mi) {
#pragma unroll
            for (int r = 0; r < 8; ++r) {
                const int m_g = m0 + mi * 16 + rb + r;
                const int bb  = m_g >> 11;        // / N_
                const int row = m_g & (N_ - 1);
                float val = (acc[mi][s][r] + bias) * scl;
                dst[((size_t)(bb * H_ + h) * N_ + row) * D_ + d] = (__bf16)val;
            }
        }
    }
}

// ---------------------------------------------------------------------------
// Kernel 2: flash attention. Grid = (N_/64 query tiles, B_*H_).
// Block = 128 threads (4 waves), each wave owns 16 query rows.
// ---------------------------------------------------------------------------
__global__ __launch_bounds__(128) void mha_flash_attn(
    const __bf16* __restrict__ q,   // [B,H,N,D], q pre-scaled
    const __bf16* __restrict__ k,   // [B,H,N,D]
    const __bf16* __restrict__ v,   // [B,H,N,D]
    __bf16* __restrict__ o) {       // [B,N,H*D]
    __shared__ __bf16 lds_v[D_][32];      // V chunk transposed: [d][key]
    __shared__ __bf16 lds_p[4][16 * 32];  // per-wave P staging (C/D -> A relayout)

    const int wave = threadIdx.x >> 5;
    const int lane = threadIdx.x & 31;
    const int bh   = blockIdx.y;
    const int bb   = bh >> 4;
    const int h    = bh & 15;

    const __bf16* qh = q + (size_t)bh * N_ * D_;
    const __bf16* kh = k + (size_t)bh * N_ * D_;
    const __bf16* vh = v + (size_t)bh * N_ * D_;
    const int q0 = blockIdx.x * 64 + wave * 16;

    // Query tile held in registers as two A operands (D = 64 = 2 x K32)
    const v16bf qa0 = load_tile16x32(qh + (size_t)q0 * D_ + 0,  D_, lane);
    const v16bf qa1 = load_tile16x32(qh + (size_t)q0 * D_ + 32, D_, lane);

    v8f o_acc[4] = {zero8(), zero8(), zero8(), zero8()};
    float m_r[8], l_r[8];
#pragma unroll
    for (int r = 0; r < 8; ++r) { m_r[r] = -__builtin_inff(); l_r[r] = 0.f; }

    const int col  = lane & 15;
    const int rb   = (lane < 16) ? 0 : 8;
    const int half = lane >> 4;

    for (int kc = 0; kc < N_; kc += 32) {
        // Prefetch next chunk's K and V lines into the cache hierarchy
        __builtin_prefetch((const void*)(kh + (size_t)(kc + 32) * D_), 0, 1);
        __builtin_prefetch((const void*)(vh + (size_t)(kc + 32) * D_), 0, 1);

        // ---- cooperatively stage V chunk transposed into LDS ----
        __syncthreads();  // previous iteration's reads are done
        {
            const int t   = threadIdx.x;       // 0..127
            const int key = t >> 2;            // 0..31
            const int dg  = (t & 3) * 16;      // 0,16,32,48
            const __bf16* src = vh + (size_t)(kc + key) * D_ + dg;
#pragma unroll
            for (int e = 0; e < 16; ++e) lds_v[dg + e][key] = src[e];
        }
        __syncthreads();

        // ---- S = q @ k^T : preload all four K B-tiles, then 4 WMMAs ----
        v8f s0 = zero8(), s1 = zero8();
        {
            v16bf kb00 = load_tile16x32(kh + (size_t)kc * D_ + 0,         D_, lane);
            v16bf kb01 = load_tile16x32(kh + (size_t)kc * D_ + 32,        D_, lane);
            v16bf kb10 = load_tile16x32(kh + (size_t)(kc + 16) * D_ + 0,  D_, lane);
            v16bf kb11 = load_tile16x32(kh + (size_t)(kc + 16) * D_ + 32, D_, lane);
            s0 = wmma_bf16(qa0, kb00, s0);
            s0 = wmma_bf16(qa1, kb01, s0);
            s1 = wmma_bf16(qa0, kb10, s1);
            s1 = wmma_bf16(qa1, kb11, s1);
        }

        // ---- online softmax: rows live in one VGPR across a 16-lane half ----
#pragma unroll
        for (int r = 0; r < 8; ++r) {
            float mx = fmaxf(s0[r], s1[r]);
            mx = fmaxf(mx, __shfl_xor(mx, 1, 32));
            mx = fmaxf(mx, __shfl_xor(mx, 2, 32));
            mx = fmaxf(mx, __shfl_xor(mx, 4, 32));
            mx = fmaxf(mx, __shfl_xor(mx, 8, 32));
            const float m_new = fmaxf(m_r[r], mx);
            const float alpha = __expf(m_r[r] - m_new);
            const float p0 = __expf(s0[r] - m_new);
            const float p1 = __expf(s1[r] - m_new);
            float rs = p0 + p1;
            rs += __shfl_xor(rs, 1, 32);
            rs += __shfl_xor(rs, 2, 32);
            rs += __shfl_xor(rs, 4, 32);
            rs += __shfl_xor(rs, 8, 32);
            l_r[r] = l_r[r] * alpha + rs;
            m_r[r] = m_new;
            s0[r] = p0; s1[r] = p1;
            o_acc[0][r] *= alpha; o_acc[1][r] *= alpha;
            o_acc[2][r] *= alpha; o_acc[3][r] *= alpha;
        }

        // ---- relayout P (C/D layout) -> A layout through per-wave LDS ----
#pragma unroll
        for (int r = 0; r < 8; ++r) {
            lds_p[wave][(rb + r) * 32 + col]      = (__bf16)s0[r];
            lds_p[wave][(rb + r) * 32 + col + 16] = (__bf16)s1[r];
        }
        asm volatile("s_wait_dscnt 0" ::: "memory");
        v16bf pa;
        {
            const int rr = lane & 15;
            v8bf lo = *(const v8bf*)&lds_p[wave][rr * 32 + half * 8];
            v8bf hi = *(const v8bf*)&lds_p[wave][rr * 32 + half * 8 + 16];
            pa = __builtin_shufflevector(lo, hi, 0, 1, 2, 3, 4, 5, 6, 7,
                                                 8, 9, 10, 11, 12, 13, 14, 15);
        }

        // ---- O += P @ V : 4 B-tiles from transposed-V LDS ----
#pragma unroll
        for (int t = 0; t < 4; ++t) {
            const int n = t * 16 + col;  // output d index (B-operand column)
            v8bf lo = *(const v8bf*)&lds_v[n][half * 8];
            v8bf hi = *(const v8bf*)&lds_v[n][half * 8 + 16];
            v16bf vb = __builtin_shufflevector(lo, hi, 0, 1, 2, 3, 4, 5, 6, 7,
                                                       8, 9, 10, 11, 12, 13, 14, 15);
            o_acc[t] = wmma_bf16(pa, vb, o_acc[t]);
        }
    }

    // ---- normalize and store to attn_out [B,N,H*D] as bf16 ----
    float inv_l[8];
#pragma unroll
    for (int r = 0; r < 8; ++r) inv_l[r] = 1.0f / l_r[r];
#pragma unroll
    for (int t = 0; t < 4; ++t) {
#pragma unroll
        for (int r = 0; r < 8; ++r) {
            const int row = q0 + rb + r;
            o[((size_t)(bb * N_ + row)) * (H_ * D_) + h * D_ + t * 16 + col] =
                (__bf16)(o_acc[t][r] * inv_l[r]);
        }
    }
}

// ---------------------------------------------------------------------------
// Kernel 3: output projection. out = attn @ w_out^T + b_out, fp32 out.
// Block tile 128x64, wave tile 32x64.
// ---------------------------------------------------------------------------
__global__ __launch_bounds__(128) void mha_out_gemm(
    const __bf16* __restrict__ ab,     // [M_, C_] bf16 attention output
    const __bf16* __restrict__ wb,     // [C_, C_] (B^T row-major)
    const float*  __restrict__ bout,   // [C_]
    float* __restrict__ out) {         // [M_, C_] fp32
    const int wave = threadIdx.x >> 5;
    const int lane = threadIdx.x & 31;
    const int m0   = blockIdx.x * 128 + wave * 32;
    const int n0   = blockIdx.y * 64;

    v8f acc[2][4] = {{zero8(), zero8(), zero8(), zero8()},
                     {zero8(), zero8(), zero8(), zero8()}};
    gemm_core_32x64(ab + (size_t)m0 * C_, wb + (size_t)n0 * C_, lane, acc);

    const int col = lane & 15;
    const int rbv = (lane < 16) ? 0 : 8;
#pragma unroll
    for (int s = 0; s < 4; ++s) {
        const int n_g = n0 + s * 16 + col;
        const float bias = bout[n_g];
#pragma unroll
        for (int mi = 0; mi < 2; ++mi) {
#pragma unroll
            for (int r = 0; r < 8; ++r) {
                const int m_g = m0 + mi * 16 + rbv + r;
                out[(size_t)m_g * C_ + n_g] = acc[mi][s][r] + bias;
            }
        }
    }
}

// ---------------------------------------------------------------------------
// Host-side launch
// ---------------------------------------------------------------------------
extern "C" void kernel_launch(void* const* d_in, const int* in_sizes, int n_in,
                              void* d_out, int out_size, void* d_ws, size_t ws_size,
                              hipStream_t stream) {
    (void)in_sizes; (void)n_in; (void)out_size; (void)ws_size;
    const float* x     = (const float*)d_in[0];  // [B,N,C]
    const float* w_qkv = (const float*)d_in[1];  // [3HD, C]
    const float* b_qkv = (const float*)d_in[2];  // [3HD]
    const float* w_out = (const float*)d_in[3];  // [C, HD]
    const float* b_out = (const float*)d_in[4];  // [C]
    float* out = (float*)d_out;

    // Workspace carve-up (bf16 elements), all offsets 16B-aligned
    char* ws = (char*)d_ws;
    size_t off = 0;
    auto carve = [&](size_t elems) -> __bf16* {
        __bf16* p = (__bf16*)(ws + off);
        off += elems * sizeof(__bf16);
        return p;
    };
    const size_t n_x    = (size_t)M_ * C_;            // 4,194,304
    const size_t n_wqkv = (size_t)QKVN * C_;          // 3,145,728
    const size_t n_wout = (size_t)C_ * C_;            // 1,048,576
    const size_t n_head = (size_t)B_ * H_ * N_ * D_;  // 4,194,304

    __bf16* xb    = carve(n_x);
    __bf16* wqkvb = carve(n_wqkv);
    __bf16* woutb = carve(n_wout);
    __bf16* qb    = carve(n_head);
    __bf16* kb    = carve(n_head);
    __bf16* vb    = carve(n_head);
    __bf16* attnb = carve(n_x);

    // 0) fp32 -> bf16 converts
    mha_cvt_bf16<<<(n_x    + 255) / 256, 256, 0, stream>>>(x,     xb,    (int)n_x);
    mha_cvt_bf16<<<(n_wqkv + 255) / 256, 256, 0, stream>>>(w_qkv, wqkvb, (int)n_wqkv);
    mha_cvt_bf16<<<(n_wout + 255) / 256, 256, 0, stream>>>(w_out, woutb, (int)n_wout);

    // 1) QKV projection (M=4096, N=3072, K=1024)
    mha_qkv_gemm<<<dim3(M_ / 128, QKVN / 64), 128, 0, stream>>>(
        xb, wqkvb, b_qkv, qb, kb, vb);

    // 2) Flash attention (32 query tiles x 32 (b,h) pairs)
    mha_flash_attn<<<dim3(N_ / 64, B_ * H_), 128, 0, stream>>>(qb, kb, vb, attnb);

    // 3) Output projection (M=4096, N=1024, K=1024)
    mha_out_gemm<<<dim3(M_ / 128, C_ / 64), 128, 0, stream>>>(attnb, woutb, b_out, out);
}
